// GCNEncoder_19834158973077
// MI455X (gfx1250) — matrix operand order repaired and checked by reading, not compiled, not run on hardware
//
#include <hip/hip_runtime.h>

// ---------------- types for WMMA (CDNA5 / gfx1250, wave32) ----------------
typedef __attribute__((ext_vector_type(16))) __bf16 v16bf;
typedef __attribute__((ext_vector_type(8)))  __bf16 v8bf;
typedef __attribute__((ext_vector_type(8)))  float  v8f;

// float -> bf16 bits, round-to-nearest-even
__device__ __forceinline__ unsigned short f2bf(float f) {
    unsigned int u = __float_as_uint(f);
    u += 0x7FFFu + ((u >> 16) & 1u);
    return (unsigned short)(u >> 16);
}

// ---------------- gcn_norm precompute ----------------
__global__ void k_init_deg(float* deg, int N) {
    int i = blockIdx.x * blockDim.x + threadIdx.x;
    if (i < N) deg[i] = 1.0f;  // self-loop weight
}

__global__ void k_deg_edges(const int* __restrict__ col, const float* __restrict__ ew,
                            float* __restrict__ deg, int E) {
    int e = blockIdx.x * blockDim.x + threadIdx.x;
    if (e < E) atomicAdd(&deg[col[e]], ew[e]);
}

__global__ void k_rsqrt(float* d, int N) {
    int i = blockIdx.x * blockDim.x + threadIdx.x;
    if (i < N) {
        float v = d[i];
        d[i] = (v > 0.0f) ? rsqrtf(v) : 0.0f;
    }
}

__global__ void k_build_norm(const int* __restrict__ row, const int* __restrict__ col,
                             const float* __restrict__ ew, const float* __restrict__ dis,
                             int* __restrict__ rc, int* __restrict__ cc,
                             float* __restrict__ nm, int E, int N) {
    int e = blockIdx.x * blockDim.x + threadIdx.x;
    if (e < E) {
        int r = row[e], c = col[e];
        rc[e] = r; cc[e] = c;
        nm[e] = dis[r] * ew[e] * dis[c];
    } else if (e < E + N) {
        int i = e - E;
        rc[e] = i; cc[e] = i;
        float d = dis[i];
        nm[e] = d * d;  // self loop: dis[i] * 1 * dis[i]
    }
}

// ---------------- conversions ----------------
// x (N x K, f32, row-major) -> hbf (Npad x LDH, bf16, zero padded)
__global__ void k_x_to_bf(const float* __restrict__ x, unsigned short* __restrict__ hbf,
                          int N, int K, int LDH, int total) {
    int i = blockIdx.x * blockDim.x + threadIdx.x;
    if (i >= total) return;
    int r = i / LDH, k = i - r * LDH;
    unsigned short v = 0;
    if (r < N && k < K) v = f2bf(x[(size_t)r * K + k]);
    hbf[i] = v;
}

// W (din x dout, f32, row-major) -> Wt (LDW x LDW bf16), Wt[n][k] = W[k][n], zero padded
__global__ void k_conv_w(const float* __restrict__ W, unsigned short* __restrict__ wt,
                         int din, int dout, int LDW) {
    int i = blockIdx.x * blockDim.x + threadIdx.x;
    if (i >= LDW * LDW) return;
    int n = i / LDW, k = i - n * LDW;
    unsigned short v = 0;
    if (n < dout && k < din) v = f2bf(W[(size_t)k * dout + n]);
    wt[i] = v;
}

// agg + bias, relu, -> bf16 zero-padded next-layer input
__global__ void k_bias_relu_bf(const float* __restrict__ agg, const float* __restrict__ bias,
                               unsigned short* __restrict__ hbf, int N, int dout,
                               int lda, int LDH, int total) {
    int i = blockIdx.x * blockDim.x + threadIdx.x;
    if (i >= total) return;
    int r = i / LDH, k = i - r * LDH;
    unsigned short v = 0;
    if (r < N && k < dout) {
        float f = agg[(size_t)r * lda + k] + bias[k];
        v = f2bf(fmaxf(f, 0.0f));
    }
    hbf[i] = v;
}

__global__ void k_bias_add(float* __restrict__ out, const float* __restrict__ b,
                           int N, int D) {
    int i = blockIdx.x * blockDim.x + threadIdx.x;
    if (i < N * D) out[i] += b[i % D];
}

// ---------------- bf16 WMMA GEMM: T[M x dout] = H[M x K] * Wt^T ----------------
// One wave computes a 16(M) x 64(N) tile: A-frag reused across 4 B-frags,
// 4x v_wmma_f32_16x16x32_bf16 per K-step. Fragment k order: contiguous, k = hi*16 + s for A and B.
__global__ __launch_bounds__(256)
void k_gemm_bf16_wmma(const unsigned short* __restrict__ Hs,
                      const unsigned short* __restrict__ Ws,
                      float* __restrict__ T,
                      int Mtiles, int NT64, int K, int ldh, int ldw, int ldt) {
    const __bf16* H  = (const __bf16*)Hs;
    const __bf16* Wt = (const __bf16*)Ws;

    int wave = (blockIdx.x * blockDim.x + threadIdx.x) >> 5;
    int lane = threadIdx.x & 31;
    if (wave >= Mtiles * NT64) return;
    int mt = wave % Mtiles;
    int nt = wave / Mtiles;

    int hi = lane >> 4;    // half-wave select
    int ln = lane & 15;

    const __bf16* arow  = H  + (size_t)(mt * 16 + ln) * ldh;          // A row m
    const __bf16* brow0 = Wt + (size_t)(nt * 64 +  0 + ln) * ldw + hi * 16;
    const __bf16* brow1 = brow0 + (size_t)16 * ldw;
    const __bf16* brow2 = brow0 + (size_t)32 * ldw;
    const __bf16* brow3 = brow0 + (size_t)48 * ldw;

    v8f acc0 = {}, acc1 = {}, acc2 = {}, acc3 = {};

    for (int k0 = 0; k0 < K; k0 += 32) {
        union { v16bf v; v8bf h[2]; } a, b0, b1, b2, b3;
        // A 16x32 bf16: lanes hold K = hi*16+[0..15]
        a.h[0] = *(const v8bf*)(arow + k0 + hi * 16);
        a.h[1] = *(const v8bf*)(arow + k0 + hi * 16 + 8);
        // B 32x16 bf16 (col n from transposed weights): K = hi*16+[0..15]
        b0.h[0] = *(const v8bf*)(brow0 + k0);
        b0.h[1] = *(const v8bf*)(brow0 + k0 + 8);
        b1.h[0] = *(const v8bf*)(brow1 + k0);
        b1.h[1] = *(const v8bf*)(brow1 + k0 + 8);
        b2.h[0] = *(const v8bf*)(brow2 + k0);
        b2.h[1] = *(const v8bf*)(brow2 + k0 + 8);
        b3.h[0] = *(const v8bf*)(brow3 + k0);
        b3.h[1] = *(const v8bf*)(brow3 + k0 + 8);

        acc0 = __builtin_amdgcn_wmma_f32_16x16x32_bf16(false, a.v, false, b0.v, (short)0, acc0, false, false);
        acc1 = __builtin_amdgcn_wmma_f32_16x16x32_bf16(false, a.v, false, b1.v, (short)0, acc1, false, false);
        acc2 = __builtin_amdgcn_wmma_f32_16x16x32_bf16(false, a.v, false, b2.v, (short)0, acc2, false, false);
        acc3 = __builtin_amdgcn_wmma_f32_16x16x32_bf16(false, a.v, false, b3.v, (short)0, acc3, false, false);
    }

    // C layout: VGPR r holds row M = hi*8 + r, col N = ln
    float* trow = T + (size_t)(mt * 16 + hi * 8) * ldt + nt * 64 + ln;
#pragma unroll
    for (int r = 0; r < 8; ++r) {
        trow[(size_t)r * ldt +  0] = acc0[r];
        trow[(size_t)r * ldt + 16] = acc1[r];
        trow[(size_t)r * ldt + 32] = acc2[r];
        trow[(size_t)r * ldt + 48] = acc3[r];
    }
}

// ---------------- sparse aggregation: agg[col] += norm * t[row] ----------------
__global__ void k_aggregate(const float* __restrict__ t, const int* __restrict__ rc,
                            const int* __restrict__ cc, const float* __restrict__ nm,
                            float* __restrict__ agg, int nE, int dout, int ldt, int lda) {
    int e = blockIdx.x;
    if (e >= nE) return;
    int r = rc[e], c = cc[e];
    float w = nm[e];
    const float* src = t + (size_t)r * ldt;
    float* dst = agg + (size_t)c * lda;
    for (int d = threadIdx.x; d < dout; d += blockDim.x)
        atomicAdd(&dst[d], w * src[d]);
}

// ---------------- launch ----------------
extern "C" void kernel_launch(void* const* d_in, const int* in_sizes, int n_in,
                              void* d_out, int out_size, void* d_ws, size_t ws_size,
                              hipStream_t stream) {
    (void)n_in; (void)out_size; (void)ws_size;

    const float* x    = (const float*)d_in[0];
    const int*   ei   = (const int*)  d_in[1];
    const float* ew   = (const float*)d_in[2];
    const float* W1   = (const float*)d_in[3];
    const float* b1   = (const float*)d_in[4];
    const float* Wmid = (const float*)d_in[5];
    const float* bmid = (const float*)d_in[6];
    const float* W7   = (const float*)d_in[7];
    const float* b7   = (const float*)d_in[8];

    const int DIN  = 128;
    const int N    = in_sizes[0] / DIN;       // 10000
    const int E    = in_sizes[1] / 2;         // 160000
    const int DH   = in_sizes[4];             // 1000
    const int DOUT = in_sizes[8];             // 256
    const int Npad = (N + 15) & ~15;          // 10000
    const int LD   = 1024;                    // padded feature stride
    const int EN   = E + N;

    // ---- workspace carve-up ----
    char* p = (char*)d_ws;
    auto take = [&](size_t bytes) -> void* {
        void* r = (void*)p;
        p += (bytes + 255) & ~(size_t)255;
        return r;
    };
    unsigned short* hbf = (unsigned short*)take((size_t)Npad * LD * 2); // bf16 activations
    unsigned short* wt  = (unsigned short*)take((size_t)LD * LD * 2);   // bf16 Wt
    float* t   = (float*)take((size_t)Npad * LD * 4);                   // GEMM output
    float* agg = (float*)take((size_t)N * LD * 4);                      // aggregation
    float* dis = (float*)take((size_t)N * 4);                           // deg -> deg^-1/2
    float* nm  = (float*)take((size_t)EN * 4);
    int*   rc  = (int*)  take((size_t)EN * 4);
    int*   cc  = (int*)  take((size_t)EN * 4);

    const int* row = ei;
    const int* col = ei + E;

    // ---- gcn_norm ----
    k_init_deg <<<(N + 255) / 256, 256, 0, stream>>>(dis, N);
    k_deg_edges<<<(E + 255) / 256, 256, 0, stream>>>(col, ew, dis, E);
    k_rsqrt    <<<(N + 255) / 256, 256, 0, stream>>>(dis, N);
    k_build_norm<<<(EN + 255) / 256, 256, 0, stream>>>(row, col, ew, dis, rc, cc, nm, E, N);

    // ---- input to bf16 padded ----
    const int htotal = Npad * LD;
    k_x_to_bf<<<(htotal + 255) / 256, 256, 0, stream>>>(x, hbf, N, DIN, LD, htotal);

    // ---- layers ----
    const float* Ws[7] = { W1,
                           Wmid + 0 * (size_t)DH * DH, Wmid + 1 * (size_t)DH * DH,
                           Wmid + 2 * (size_t)DH * DH, Wmid + 3 * (size_t)DH * DH,
                           Wmid + 4 * (size_t)DH * DH, W7 };
    const float* bs[7] = { b1, bmid, bmid + DH, bmid + 2 * DH, bmid + 3 * DH, bmid + 4 * DH, b7 };
    int dins[7]  = { DIN, DH, DH, DH, DH, DH, DH };
    int douts[7] = { DH,  DH, DH, DH, DH, DH, DOUT };

    const int Mtiles = Npad / 16;

    for (int l = 0; l < 7; ++l) {
        const int din = dins[l], dout = douts[l];

        // transpose + convert weights (zero padded to LD x LD)
        k_conv_w<<<(LD * LD + 255) / 256, 256, 0, stream>>>(Ws[l], wt, din, dout, LD);

        // GEMM: t = hbf @ W  (K rounded up to 32; padding regions are zero)
        const int Kiter = (din + 31) & ~31;
        const int NT64  = ((dout + 63) & ~63) / 64;
        const int waves = Mtiles * NT64;
        k_gemm_bf16_wmma<<<(waves * 32 + 255) / 256, 256, 0, stream>>>(
            hbf, wt, t, Mtiles, NT64, Kiter, LD, LD, LD);

        // aggregation target
        const bool last = (l == 6);
        float* ad  = last ? (float*)d_out : agg;
        const int lda = last ? DOUT : LD;
        const size_t abytes = last ? (size_t)N * DOUT * 4 : (size_t)N * LD * 4;
        hipMemsetAsync(ad, 0, abytes, stream);

        k_aggregate<<<EN, 256, 0, stream>>>(t, rc, cc, nm, ad, EN, dout, LD, lda);

        if (!last) {
            k_bias_relu_bf<<<(htotal + 255) / 256, 256, 0, stream>>>(
                agg, bs[l], hbf, N, dout, LD, LD, htotal);
        } else {
            k_bias_add<<<(N * DOUT + 255) / 256, 256, 0, stream>>>(
                (float*)d_out, bs[l], N, DOUT);
        }
    }
}
